// TwoSimplicialAttention_82454782148619
// MI455X (gfx1250) — matrix operand order repaired and checked by reading, not compiled, not run on hardware
//
#include <hip/hip_runtime.h>
#include <hip/hip_bf16.h>

typedef __attribute__((ext_vector_type(16))) __bf16 v16bf;
typedef __attribute__((ext_vector_type(8)))  float  v8f;
typedef __attribute__((ext_vector_type(4)))  unsigned int v4u;
typedef __attribute__((ext_vector_type(8)))  int v8i;
typedef __attribute__((ext_vector_type(4)))  int v4i;
typedef unsigned short u16t;

#define DMODEL 1024
#define SEQLEN 2048
#define NBATCH 2
#define NHEADS 16
#define HDIM   64
#define MROWS  (NBATCH*SEQLEN)   /* 4096 */
#define WIN    16                /* W1 == W2 == 16 */

#if defined(__HIP_DEVICE_COMPILE__) && __has_builtin(__builtin_amdgcn_tensor_load_to_lds)
#define USE_TDM 1
#else
#define USE_TDM 0
#endif

// ---------- WMMA fragment loaders (per CDNA5 ISA layouts) ----------
// A 16x32 bf16: lane<16 -> row lm, K {0..7}u{16..23}; lane>=16 -> K {8..15}u{24..31}.
static __device__ __forceinline__ v16bf load_frag_a(const __bf16* rowbase, int hi) {
    union { uint4 u[2]; v16bf v; } r;
    const char* p = (const char*)rowbase;
    r.u[0] = *(const uint4*)(p + hi * 16);
    r.u[1] = *(const uint4*)(p + 32 + hi * 16);
    return r.v;
}
// B 32x16 bf16: lane<16 -> col lm, K {0..7}u{16..23}; lane>=16 -> K {8..15}u{24..31}.
static __device__ __forceinline__ v16bf load_frag_b(const __bf16* rowbase, int hi) {
    union { uint4 u[2]; v16bf v; } r;
    const char* p = (const char*)rowbase;
    r.u[0] = *(const uint4*)(p + hi * 16);
    r.u[1] = *(const uint4*)(p + 32 + hi * 16);
    return r.v;
}

#if USE_TDM
// Tensor DMA descriptor per CDNA5 ISA (2D tile).
// group0: count=1 | lds_addr | global_addr(57b) | type=2
// group1: data_size=1 (2B), tensor_dim0/1, tile_dim0/1, tensor_dim0_stride
static __device__ __forceinline__ void tdm_load_2d(
    unsigned lds_addr, const void* gaddr,
    unsigned tensor_d0, unsigned tensor_d1, unsigned stride0,
    unsigned tile_d0, unsigned tile_d1) {
    unsigned long long ga = (unsigned long long)(uintptr_t)gaddr;
    v4u g0;
    g0.x = 1u;                                         // count=1 (valid), user mode
    g0.y = lds_addr;                                   // bytes
    g0.z = (unsigned)(ga & 0xffffffffu);               // global_addr[31:0]
    g0.w = (unsigned)((ga >> 32) & 0x01ffffffu) | (2u << 30);  // addr[56:32] | type=2
    v8i g1;
    g1[0] = (int)(1u << 16);                           // workgroup_mask=0, data_size=1 (2B)
    g1[1] = (int)((tensor_d0 & 0xffffu) << 16);        // tensor_dim0[15:0] @bits63:48
    g1[2] = (int)(((tensor_d0 >> 16) & 0xffffu) | ((tensor_d1 & 0xffffu) << 16));
    g1[3] = (int)(((tensor_d1 >> 16) & 0xffffu) | ((tile_d0 & 0xffffu) << 16));
    g1[4] = (int)(tile_d1 & 0xffffu);                  // tile_dim1 | tile_dim2=0
    g1[5] = (int)stride0;                              // tensor_dim0_stride[31:0]
    g1[6] = 0;                                         // stride0 hi | stride1 lo
    g1[7] = 0;
    v4i z4 = {0, 0, 0, 0};                             // groups 2/3: dims/tiles unused
    v8i z8 = {0, 0, 0, 0, 0, 0, 0, 0};
    __builtin_amdgcn_tensor_load_to_lds(g0, g1, z4, z4, z8, 0);  // 6-arg toolchain form
}
#endif

// ---------- x fp32 -> bf16 ----------
__global__ __launch_bounds__(256) void cvt_x_kernel(const float* __restrict__ x,
                                                    __bf16* __restrict__ xb) {
    int i = (blockIdx.x * 256 + threadIdx.x) * 4;
    float4 v = *(const float4*)(x + i);
    union { __bf16 b[4]; uint2 u; } o;
    o.b[0] = (__bf16)v.x; o.b[1] = (__bf16)v.y;
    o.b[2] = (__bf16)v.z; o.b[3] = (__bf16)v.w;
    *(uint2*)(xb + i) = o.u;
}

// ---------- W (K x N fp32) -> WT (N x K bf16) ----------
__global__ __launch_bounds__(256) void transpose_cvt_kernel(
    const float* W0, const float* W1, const float* W2,
    const float* W3, const float* W4, const float* W5,
    __bf16* T0, __bf16* T1, __bf16* T2, __bf16* T3, __bf16* T4, __bf16* T5) {
    __shared__ float tile[32][33];
    int z = blockIdx.z;
    const float* W = (z == 0) ? W0 : (z == 1) ? W1 : (z == 2) ? W2
                   : (z == 3) ? W3 : (z == 4) ? W4 : W5;
    __bf16* T = (z == 0) ? T0 : (z == 1) ? T1 : (z == 2) ? T2
              : (z == 3) ? T3 : (z == 4) ? T4 : T5;
    int tid = threadIdx.x;
    int tx = tid & 31, ty = tid >> 5;
    int n0 = blockIdx.x * 32, k0 = blockIdx.y * 32;
    for (int i = 0; i < 4; i++) {
        int k = ty + i * 8;
        tile[k][tx] = W[(size_t)(k0 + k) * DMODEL + n0 + tx];
    }
    __syncthreads();
    for (int i = 0; i < 4; i++) {
        int n = ty + i * 8;
        T[(size_t)(n0 + n) * DMODEL + k0 + tx] = (__bf16)tile[tx][n];
    }
}

// ---------- inner WMMA compute on one BK=64 LDS tile pair ----------
static __device__ __forceinline__ void gemm_tile_compute(
    const __bf16* as, const __bf16* bs, int wr, int wc, int lm, int hi,
    v8f acc[2][2]) {
    for (int kc = 0; kc < 2; kc++) {
        v16bf af[2], bfr[2];
        for (int i = 0; i < 2; i++)
            af[i] = load_frag_a(as + (wr * 32 + i * 16 + lm) * 64 + kc * 32, hi);
        for (int j = 0; j < 2; j++)
            bfr[j] = load_frag_b(bs + (wc * 32 + j * 16 + lm) * 64 + kc * 32, hi);
        for (int i = 0; i < 2; i++)
            for (int j = 0; j < 2; j++)
                acc[i][j] = __builtin_amdgcn_wmma_f32_16x16x32_bf16(
                    false, af[i], false, bfr[j], (short)0, acc[i][j], false, false);
    }
}

// ---------- bf16 WMMA GEMM: out = A(MxK) @ W(KxN) + bias (W given as WT = N x K) ----
// BM=128, BN=64, BK=64; TDM double-buffered LDS; 8 waves each a 32x32 subtile.
template <bool OUTB>
__global__ __launch_bounds__(256) void gemm_bf16_kernel(
    const __bf16* __restrict__ A, const __bf16* __restrict__ WT,
    const float* __restrict__ bias,
    __bf16* __restrict__ outb, float* __restrict__ outf,
    const float* __restrict__ resid, int M, int N, int K) {
    __shared__ __bf16 As[2][128 * 64];   // 2 x 16 KB
    __shared__ __bf16 Bs[2][64 * 64];    // 2 x  8 KB
    const int tid = threadIdx.x;
    const int lane = tid & 31, wave = tid >> 5;
    const int lm = lane & 15, hi = lane >> 4;
    const int wr = wave & 3, wc = wave >> 2;
    const int gm = blockIdx.x * 128, gn = blockIdx.y * 64;

    v8f acc[2][2];
    for (int i = 0; i < 2; i++)
        for (int j = 0; j < 2; j++)
            for (int r = 0; r < 8; r++) acc[i][j][r] = 0.f;

#if USE_TDM
    const unsigned asb = (unsigned)(uintptr_t)(void*)&As[0][0];
    const unsigned bsb = (unsigned)(uintptr_t)(void*)&Bs[0][0];
    if (wave == 0) {
        tdm_load_2d(asb, A + (size_t)gm * K, K, M, K, 64, 128);
        tdm_load_2d(bsb, WT + (size_t)gn * K, K, N, K, 64, 64);
    }
    int buf = 0;
    for (int kb = 0; kb < K; kb += 64) {
        const bool more = (kb + 64) < K;
        if (wave == 0) {
            if (more) {
                tdm_load_2d(asb + (buf ^ 1) * (128 * 64 * 2),
                            A + (size_t)gm * K + kb + 64, K, M, K, 64, 128);
                tdm_load_2d(bsb + (buf ^ 1) * (64 * 64 * 2),
                            WT + (size_t)gn * K + kb + 64, K, N, K, 64, 64);
                __builtin_amdgcn_s_wait_tensorcnt(2);  // prev pair done, next in flight
            } else {
                __builtin_amdgcn_s_wait_tensorcnt(0);
            }
        }
        __syncthreads();                                // tile[buf] visible to all waves
        gemm_tile_compute(As[buf], Bs[buf], wr, wc, lm, hi, acc);
        __syncthreads();                                // reads done before TDM overwrite
        buf ^= 1;
    }
#else
    const int arow = tid >> 1, ahalf = tid & 1;         // 128 rows x 2 (64B) halves
    const int brow = tid >> 2, bq = tid & 3;            // 64 rows x 4 (32B) chunks
    for (int kb = 0; kb < K; kb += 64) {
        const uint4* asrc = (const uint4*)(A + (size_t)(gm + arow) * K + kb + ahalf * 32);
        uint4* adst = (uint4*)(&As[0][arow * 64 + ahalf * 32]);
        adst[0] = asrc[0]; adst[1] = asrc[1]; adst[2] = asrc[2]; adst[3] = asrc[3];
        const uint4* bsrc = (const uint4*)(WT + (size_t)(gn + brow) * K + kb + bq * 16);
        uint4* bdst = (uint4*)(&Bs[0][brow * 64 + bq * 16]);
        bdst[0] = bsrc[0]; bdst[1] = bsrc[1];
        __syncthreads();
        gemm_tile_compute(As[0], Bs[0], wr, wc, lm, hi, acc);
        __syncthreads();
    }
#endif

    for (int j = 0; j < 2; j++) {
        int n = gn + wc * 32 + j * 16 + lm;
        float bv = bias[n];
        for (int i = 0; i < 2; i++) {
            int mbase = gm + wr * 32 + i * 16 + hi * 8;
            for (int r = 0; r < 8; r++) {
                size_t off = (size_t)(mbase + r) * N + n;
                float v = acc[i][j][r] + bv;
                if constexpr (OUTB) {
                    outb[off] = (__bf16)v;
                } else {
                    outf[off] = v + resid[off];
                }
            }
        }
    }
}

// ---------- attention: one wave per (b, s, h) ----------
__global__ __launch_bounds__(256) void attn_kernel(
    const __bf16* __restrict__ Qb, const __bf16* __restrict__ Kb,
    const __bf16* __restrict__ Vb, const __bf16* __restrict__ Kpb,
    const __bf16* __restrict__ Vpb, __bf16* __restrict__ Zb) {
    __shared__ __align__(16) u16t smem[8 * 256];
    const int tid = threadIdx.x;
    const int lane = tid & 31, widx = tid >> 5;
    const int lm = lane & 15, hi = lane >> 4;
    const int gw = blockIdx.x * 8 + widx;
    const int h = gw & 15;
    const int s = (gw >> 4) & (SEQLEN - 1);
    const int b = gw >> 15;
    const size_t bbase = (size_t)b * SEQLEN * DMODEL;
    const int hoff = h * HDIM;

    const __bf16* qrow = Qb + bbase + (size_t)s * DMODEL + hoff;
    int jrow = s - (WIN - 1) + lm; if (jrow < 0) jrow = 0;
    const __bf16* krow  = Kb  + bbase + (size_t)jrow * DMODEL + hoff;
    const __bf16* kprow = Kpb + bbase + (size_t)jrow * DMODEL + hoff;

    // prefetch the scattered value rows needed in phase 2 (global_prefetch_b8)
    __builtin_prefetch(Vb  + bbase + (size_t)jrow * DMODEL + hoff, 0, 1);
    __builtin_prefetch(Vpb + bbase + (size_t)jrow * DMODEL + hoff, 0, 1);

    // --- phase 1: A[j,k'] = sum_d (Kw[j,d]*Q[d]) * Kpw[k',d] ---
    v8f sacc;
    for (int r = 0; r < 8; r++) sacc[r] = 0.f;
    for (int c = 0; c < 2; c++) {
        v16bf qf = load_frag_a(qrow + c * 32, hi);
        v16bf kf = load_frag_a(krow + c * 32, hi);
        v16bf tf;
        for (int i = 0; i < 16; i++) tf[i] = (__bf16)((float)kf[i] * (float)qf[i]);
        v16bf bf2 = load_frag_b(kprow + c * 32, hi);
        sacc = __builtin_amdgcn_wmma_f32_16x16x32_bf16(
            false, tf, false, bf2, (short)0, sacc, false, false);
    }

    // --- mask + scale + joint softmax over the wave-held 16x16 tile ---
    float a[8];
    for (int r = 0; r < 8; r++) {
        int j = r + 8 * hi;
        bool valid = (j >= (WIN - 1) - s) && (lm >= (WIN - 1) - s);
        a[r] = valid ? sacc[r] * 0.125f : -1e30f;
    }
    float mx = a[0];
    for (int r = 1; r < 8; r++) mx = fmaxf(mx, a[r]);
    for (int off = 16; off; off >>= 1) mx = fmaxf(mx, __shfl_xor(mx, off));
    float e[8], sum = 0.f;
    for (int r = 0; r < 8; r++) { e[r] = __expf(a[r] - mx); sum += e[r]; }
    for (int off = 16; off; off >>= 1) sum += __shfl_xor(sum, off);
    float inv = 1.f / sum;

    u16t* my = smem + widx * 256;
    for (int r = 0; r < 8; r++) {
        __bf16 v = (__bf16)(e[r] * inv);
        my[(r + 8 * hi) * 16 + lm] = __builtin_bit_cast(u16t, v);
    }
    __syncthreads();

    union Fu { uint4 u[2]; v16bf v; };
    Fu a2;
    a2.u[0] = *(const uint4*)((const char*)my + lm * 32 + hi * 16);
    a2.u[1].x = 0u; a2.u[1].y = 0u; a2.u[1].z = 0u; a2.u[1].w = 0u;

    // --- phase 2: U = Sm @ Vpw (K padded to 32), Z[d] = sum_j Vw[j,d]*U[j,d] ---
    for (int c = 0; c < 4; c++) {
        int d = c * 16 + lm;
        Fu b2;
        for (int i = 0; i < 2; i++) { b2.u[i].x = 0u; b2.u[i].y = 0u; b2.u[i].z = 0u; b2.u[i].w = 0u; }
        {
            for (int k = 0; k < 8; k++) {
                int sr = s - (WIN - 1) + 8 * hi + k; if (sr < 0) sr = 0;
                b2.v[k] = Vpb[bbase + (size_t)sr * DMODEL + hoff + d];
            }
        }
        v8f uacc;
        for (int r = 0; r < 8; r++) uacc[r] = 0.f;
        uacc = __builtin_amdgcn_wmma_f32_16x16x32_bf16(
            false, a2.v, false, b2.v, (short)0, uacc, false, false);

        float part = 0.f;
        for (int r = 0; r < 8; r++) {
            int j = r + 8 * hi;
            int sr = s - (WIN - 1) + j; if (sr < 0) sr = 0;
            part += (float)Vb[bbase + (size_t)sr * DMODEL + hoff + d] * uacc[r];
        }
        float tot = part + __shfl_xor(part, 16);
        if (hi == 0)
            Zb[bbase + (size_t)s * DMODEL + hoff + d] = (__bf16)tot;
    }
}

// ---------- LayerNorm (in-place on d_out) ----------
__global__ __launch_bounds__(256) void ln_kernel(float* __restrict__ out,
                                                 const float* __restrict__ gamma,
                                                 const float* __restrict__ beta) {
    __shared__ float red[16];
    int tid = threadIdx.x;
    float* p = out + (size_t)blockIdx.x * DMODEL;
    float v[4], sum = 0.f, sq = 0.f;
    for (int i = 0; i < 4; i++) {
        v[i] = p[tid + i * 256];
        sum += v[i]; sq += v[i] * v[i];
    }
    for (int off = 16; off; off >>= 1) { sum += __shfl_xor(sum, off); sq += __shfl_xor(sq, off); }
    int wv = tid >> 5;
    if ((tid & 31) == 0) { red[wv] = sum; red[8 + wv] = sq; }
    __syncthreads();
    float ts = 0.f, tq = 0.f;
    for (int w = 0; w < 8; w++) { ts += red[w]; tq += red[8 + w]; }
    float mu = ts * (1.f / DMODEL);
    float var = tq * (1.f / DMODEL) - mu * mu;
    float rinv = rsqrtf(var + 1e-5f);
    for (int i = 0; i < 4; i++) {
        int c = tid + i * 256;
        p[c] = (v[i] - mu) * rinv * gamma[c] + beta[c];
    }
}

extern "C" void kernel_launch(void* const* d_in, const int* in_sizes, int n_in,
                              void* d_out, int out_size, void* d_ws, size_t ws_size,
                              hipStream_t stream) {
    (void)in_sizes; (void)n_in; (void)out_size; (void)ws_size;
    const float* x   = (const float*)d_in[0];
    const float* Wq  = (const float*)d_in[1];  const float* bq  = (const float*)d_in[2];
    const float* Wk  = (const float*)d_in[3];  const float* bk  = (const float*)d_in[4];
    const float* Wv  = (const float*)d_in[5];  const float* bv  = (const float*)d_in[6];
    const float* Wkp = (const float*)d_in[7];  const float* bkp = (const float*)d_in[8];
    const float* Wvp = (const float*)d_in[9];  const float* bvp = (const float*)d_in[10];
    const float* Wo  = (const float*)d_in[11]; const float* bo  = (const float*)d_in[12];
    const float* gamma = (const float*)d_in[13];
    const float* beta  = (const float*)d_in[14];
    float* out = (float*)d_out;

    const size_t MK = (size_t)MROWS * DMODEL;
    const size_t KK = (size_t)DMODEL * DMODEL;
    __bf16* ws = (__bf16*)d_ws;
    __bf16* xb  = ws;
    __bf16* WT0 = ws + MK;
    __bf16* WT1 = WT0 + KK;  __bf16* WT2 = WT1 + KK;
    __bf16* WT3 = WT2 + KK;  __bf16* WT4 = WT3 + KK;  __bf16* WT5 = WT4 + KK;
    __bf16* Qb  = WT5 + KK;
    __bf16* Kb  = Qb + MK;   __bf16* Vb  = Kb + MK;
    __bf16* Kpb = Vb + MK;   __bf16* Vpb = Kpb + MK;
    __bf16* Zb  = Vpb + MK;                      // total 68 MB of workspace

    cvt_x_kernel<<<(int)(MK / 1024), 256, 0, stream>>>(x, xb);
    transpose_cvt_kernel<<<dim3(32, 32, 6), 256, 0, stream>>>(
        Wq, Wk, Wv, Wkp, Wvp, Wo, WT0, WT1, WT2, WT3, WT4, WT5);

    dim3 gg(MROWS / 128, DMODEL / 64);
    gemm_bf16_kernel<true><<<gg, 256, 0, stream>>>(xb, WT0, bq,  Qb,  nullptr, nullptr, MROWS, DMODEL, DMODEL);
    gemm_bf16_kernel<true><<<gg, 256, 0, stream>>>(xb, WT1, bk,  Kb,  nullptr, nullptr, MROWS, DMODEL, DMODEL);
    gemm_bf16_kernel<true><<<gg, 256, 0, stream>>>(xb, WT2, bv,  Vb,  nullptr, nullptr, MROWS, DMODEL, DMODEL);
    gemm_bf16_kernel<true><<<gg, 256, 0, stream>>>(xb, WT3, bkp, Kpb, nullptr, nullptr, MROWS, DMODEL, DMODEL);
    gemm_bf16_kernel<true><<<gg, 256, 0, stream>>>(xb, WT4, bvp, Vpb, nullptr, nullptr, MROWS, DMODEL, DMODEL);

    attn_kernel<<<(NBATCH * SEQLEN * NHEADS) / 8, 256, 0, stream>>>(Qb, Kb, Vb, Kpb, Vpb, Zb);

    gemm_bf16_kernel<false><<<gg, 256, 0, stream>>>(Zb, WT5, bo, nullptr, out, x, MROWS, DMODEL, DMODEL);
    ln_kernel<<<MROWS, 256, 0, stream>>>(out, gamma, beta);
}
